// StackedRNNModel_5308579578439
// MI455X (gfx1250) — matrix operand order repaired and checked by reading, not compiled, not run on hardware
//
#include <hip/hip_runtime.h>

typedef __attribute__((ext_vector_type(16))) __bf16 bf16x16;
typedef __attribute__((ext_vector_type(8)))  __bf16 bf16x8;
typedef __attribute__((ext_vector_type(8)))  float  v8f;

#define DEVINL __device__ __forceinline__

// ---- sizing ----
#define WAVES_PER_BLOCK 8
#define NBLOCKS 16                      // 16 blocks * 8 waves * 16 rows = 2048 samples
#define HALVES_BIG   (10 * 16384)       // 10 pre-swizzled 128x128 bf16 matrices
#define HALVES_WI0   (1 * 8 * 512)      // Wi_0 padded to 32x128
#define HALVES_WI1   (5 * 8 * 512)      // Wi_1 padded to 160x128
#define HALVES_WDEC  (4 * 512)          // Wdec padded to 128x16 (one n-tile)
// 2 layers * 8 rows * (16x128) bf16  +  one always-zero slot
#define STATE_HALVES_PER_WAVE (2 * 8 * 2048 + 2048)

// B-fragment offset for element (k, n) of a K x 128 matrix, in this file's contiguous k order (as loadAfragLDS)
// (16-bit B matrix, wave32): lane = n%16 + 16*(k%32>=16), half = k%16
DEVINL int fragoff(int k, int n) {
  int lane = (n & 15) | (((k >> 4) & 1) << 4);
  int half = k & 15;
  return ((k >> 5) * 8 + (n >> 4)) * 512 + lane * 16 + half;
}

// ======================= weight pre-swizzle kernel =======================
struct WInit {
  const float* big[10];
  const float* wi0src;
  const float* wi1src;
  const float* wdecsrc;
  __bf16* wbig;
  __bf16* wi0;
  __bf16* wi1;
  __bf16* wdec;
};

__global__ void init_weights_kernel(WInit p) {
  int tid = blockIdx.x * blockDim.x + threadIdx.x;
  if (tid < HALVES_BIG) {
    int mi = tid >> 14, e = tid & 16383;
    int k = e >> 7, n = e & 127;
    p.wbig[mi * 16384 + fragoff(k, n)] = (__bf16)p.big[mi][e];
  } else if (tid < HALVES_BIG + HALVES_WI0) {
    int e = tid - HALVES_BIG;
    int k = e >> 7, n = e & 127;
    float v = (k < 4) ? p.wi0src[k * 128 + n] : 0.0f;
    p.wi0[fragoff(k, n)] = (__bf16)v;
  } else if (tid < HALVES_BIG + HALVES_WI0 + HALVES_WI1) {
    int e = tid - HALVES_BIG - HALVES_WI0;
    int k = e >> 7, n = e & 127;   // k in [0,160)
    float v = (k < 130) ? p.wi1src[k * 128 + n] : 0.0f;
    p.wi1[fragoff(k, n)] = (__bf16)v;
  } else if (tid < HALVES_BIG + HALVES_WI0 + HALVES_WI1 + HALVES_WDEC) {
    int e = tid - HALVES_BIG - HALVES_WI0 - HALVES_WI1;   // [0, 2048)
    int k = e >> 4, n = e & 15;                           // 128 x 16
    float v = (n < 2) ? p.wdecsrc[k * 2 + n] : 0.0f;
    p.wdec[fragoff(k, n)] = (__bf16)v;
  }
}

// ======================= device helpers =======================
DEVINL void waitds() { asm volatile("s_wait_dscnt 0" ::: "memory"); }

DEVINL bf16x16 cat8(bf16x8 lo, bf16x8 hi) {
  return __builtin_shufflevector(lo, hi, 0,1,2,3,4,5,6,7,8,9,10,11,12,13,14,15);
}

DEVINL bf16x16 zero16() {
  bf16x16 z;
#pragma unroll
  for (int t = 0; t < 16; ++t) z[t] = (__bf16)0.0f;
  return z;
}

// B fragment: pre-swizzled global memory, 32B contiguous per lane (2x b128)
DEVINL bf16x16 loadBfrag(const __bf16* base, int kt, int nt, int lane) {
  const __bf16* p = base + (((kt << 3) + nt) << 9) + (lane << 4);
  return cat8(*(const bf16x8*)p, *(const bf16x8*)(p + 8));
}

// A fragment from row-major bf16 LDS stage (16 rows x 128 cols), contiguous k = 16*(lane/16) + s
DEVINL bf16x16 loadAfragLDS(const __bf16* stg, int kt, int lane) {
  int m   = lane & 15;
  int sel = (lane < 16) ? 0 : 16;
  const __bf16* p = stg + m * 128 + kt * 32 + sel;
  return cat8(*(const bf16x8*)p, *(const bf16x8*)(p + 8));
}

// A fragment stored flat per (kt, lane) in global state
DEVINL bf16x16 loadAfragG(const __bf16* slot, int kt, int lane) {
  const __bf16* p = slot + (kt << 9) + (lane << 4);
  return cat8(*(const bf16x8*)p, *(const bf16x8*)(p + 8));
}
DEVINL void storeAfragG(__bf16* slot, int kt, int lane, bf16x16 f) {
  __bf16* p = slot + (kt << 9) + (lane << 4);
  *(bf16x8*)p       = __builtin_shufflevector(f, f, 0,1,2,3,4,5,6,7);
  *(bf16x8*)(p + 8) = __builtin_shufflevector(f, f, 8,9,10,11,12,13,14,15);
}

DEVINL v8f wmma_bf16(bf16x16 a, bf16x16 b, v8f c) {
  return __builtin_amdgcn_wmma_f32_16x16x32_bf16(false, a, false, b, (short)0, c, false, false);
}

// write f32 C-layout accumulators (8 n-tiles) to LDS stage as bf16 row-major 16x128
DEVINL void stageC(__bf16* stg, v8f (&acc)[8], int lane) {
  int n  = lane & 15;
  int mb = (lane < 16) ? 0 : 8;
#pragma unroll
  for (int nt = 0; nt < 8; ++nt) {
#pragma unroll
    for (int r = 0; r < 8; ++r)
      stg[(mb + r) * 128 + nt * 16 + n] = (__bf16)acc[nt][r];
  }
}

DEVINL float sigmoidf_(float x) { return 1.0f / (1.0f + __expf(-x)); }

#if __has_builtin(__builtin_amdgcn_tanhf)
DEVINL float tanhf_(float x) { return __builtin_amdgcn_tanhf(x); }
#elif __has_builtin(__builtin_amdgcn_tanh_f32)
DEVINL float tanhf_(float x) { return __builtin_amdgcn_tanh_f32(x); }
#else
DEVINL float tanhf_(float x) { return 1.0f - 2.0f / (1.0f + __expf(2.0f * x)); }
#endif

// ---- software-pipelined GEMM phases (load B fragment t+1 before WMMA t) ----

// acc += h1@U1 + h2@U2 (dual A, dual rotated B streams)
DEVINL void gemmUU(v8f (&acc)[8], const __bf16* slotL, const __bf16* slotD,
                   const __bf16* U1, const __bf16* U2, int lane) {
#pragma unroll
  for (int kt = 0; kt < 4; ++kt) {
    bf16x16 h1 = loadAfragG(slotL, kt, lane);
    bf16x16 h2 = loadAfragG(slotD, kt, lane);
    bf16x16 b1 = loadBfrag(U1, kt, 0, lane);
    bf16x16 b2 = loadBfrag(U2, kt, 0, lane);
#pragma unroll
    for (int nt = 0; nt < 8; ++nt) {
      bf16x16 n1 = b1, n2 = b2;
      if (nt + 1 < 8) {
        n1 = loadBfrag(U1, kt, nt + 1, lane);
        n2 = loadBfrag(U2, kt, nt + 1, lane);
      }
      acc[nt] = wmma_bf16(h1, b1, acc[nt]);
      acc[nt] = wmma_bf16(h2, b2, acc[nt]);
      b1 = n1; b2 = n2;
    }
  }
}

// acc += A @ B (KT k-steps, rotated single B stream)
template <int KT>
DEVINL void gemm_acc(v8f (&acc)[8], const bf16x16 (&A)[KT], const __bf16* B, int lane) {
  bf16x16 b = loadBfrag(B, 0, 0, lane);
#pragma unroll
  for (int t = 0; t < KT * 8; ++t) {
    bf16x16 nb = b;
    if (t + 1 < KT * 8) nb = loadBfrag(B, (t + 1) >> 3, (t + 1) & 7, lane);
    acc[t & 7] = wmma_bf16(A[t >> 3], b, acc[t & 7]);
    b = nb;
  }
}

// a1 += A@B1 ; a2 += A@B2 (dual rotated B streams, shared A)
DEVINL void gemm2_acc(v8f (&a1)[8], v8f (&a2)[8], const bf16x16 (&A)[4],
                      const __bf16* B1, const __bf16* B2, int lane) {
  bf16x16 b1 = loadBfrag(B1, 0, 0, lane);
  bf16x16 b2 = loadBfrag(B2, 0, 0, lane);
#pragma unroll
  for (int t = 0; t < 32; ++t) {
    bf16x16 n1 = b1, n2 = b2;
    if (t + 1 < 32) {
      n1 = loadBfrag(B1, (t + 1) >> 3, (t + 1) & 7, lane);
      n2 = loadBfrag(B2, (t + 1) >> 3, (t + 1) & 7, lane);
    }
    a1[t & 7] = wmma_bf16(A[t >> 3], b1, a1[t & 7]);
    a2[t & 7] = wmma_bf16(A[t >> 3], b2, a2[t & 7]);
    b1 = n1; b2 = n2;
  }
}

struct LayerW {
  const __bf16 *U1, *U2, *Wh, *Wo1, *Wo2, *Wi;
  const float  *bi, *bo1, *bo2;   // staged in LDS
};

// One RNN layer for one 16-row batch tile. Leaves x = o1*sigmoid(o2) staged (bf16) in stg.
template <int NKIN>
DEVINL void run_layer(const LayerW& w, const __bf16* slot_left, __bf16* slot_cur,
                      const bf16x16 (&inF)[NKIN], __bf16* stg, int lane) {
  v8f acc[8];
  // ---- h = h1@U1 + h2@U2 (slot_left may be the always-zero slot) ----
#pragma unroll
  for (int nt = 0; nt < 8; ++nt)
#pragma unroll
    for (int r = 0; r < 8; ++r) acc[nt][r] = 0.0f;
  gemmUU(acc, slot_left, slot_cur, w.U1, w.U2, lane);
  // ---- relayout h to A fragments via LDS ----
  stageC(stg, acc, lane);
  waitds();
  bf16x16 hA[4];
#pragma unroll
  for (int kt = 0; kt < 4; ++kt) hA[kt] = loadAfragLDS(stg, kt, lane);
  waitds();
  // ---- pre-activation = cat(x1,x2)@Wi + bi + h@Wh ----
#pragma unroll
  for (int nt = 0; nt < 8; ++nt) {
    float bv = w.bi[nt * 16 + (lane & 15)];
#pragma unroll
    for (int r = 0; r < 8; ++r) acc[nt][r] = bv;
  }
  gemm_acc<NKIN>(acc, inF, w.Wi, lane);
  gemm_acc<4>(acc, hA, w.Wh, lane);
  // ---- tanh ----
#pragma unroll
  for (int nt = 0; nt < 8; ++nt)
#pragma unroll
    for (int r = 0; r < 8; ++r) acc[nt][r] = tanhf_(acc[nt][r]);
  // ---- h_new -> A frags, persist to state ----
  stageC(stg, acc, lane);
  waitds();
  bf16x16 hn[4];
#pragma unroll
  for (int kt = 0; kt < 4; ++kt) {
    hn[kt] = loadAfragLDS(stg, kt, lane);
    storeAfragG(slot_cur, kt, lane, hn[kt]);
  }
  waitds();
  // ---- x = (h_new@Wo1 + bo1) * sigmoid(h_new@Wo2 + bo2) ----
  v8f a2[8];
#pragma unroll
  for (int nt = 0; nt < 8; ++nt) {
    float b1 = w.bo1[nt * 16 + (lane & 15)];
    float b2 = w.bo2[nt * 16 + (lane & 15)];
#pragma unroll
    for (int r = 0; r < 8; ++r) { acc[nt][r] = b1; a2[nt][r] = b2; }
  }
  gemm2_acc(acc, a2, hn, w.Wo1, w.Wo2, lane);
#pragma unroll
  for (int nt = 0; nt < 8; ++nt)
#pragma unroll
    for (int r = 0; r < 8; ++r) acc[nt][r] *= sigmoidf_(a2[nt][r]);
  stageC(stg, acc, lane);
  waitds();
}

// ======================= main kernel =======================
struct MainP {
  const int* samples;
  const __bf16 *wbig, *wi0, *wi1, *wdec;
  __bf16* state;
  const float *bi0, *bo10, *bo20, *bi1, *bo11, *bo21;
  const float *bdec, *Wph, *bph;
  float* out;
};

__global__ __launch_bounds__(256) void rnn_kernel(MainP p) {
  __shared__ __bf16 stg_all[WAVES_PER_BLOCK][2048];       // 16x128 bf16 per wave
  __shared__ float  ohc_all[WAVES_PER_BLOCK][8][16][2];   // one-hot column state
  __shared__ float  lgb_all[WAVES_PER_BLOCK][16][2];      // per-row logits exchange
  __shared__ float  bias_all[6][128];                     // bi0,bo10,bo20,bi1,bo11,bo21

  const int lane = threadIdx.x & 31;
  const int wv   = threadIdx.x >> 5;
  const int wid  = blockIdx.x * WAVES_PER_BLOCK + wv;
  const int m0   = wid * 16;
  const int m    = lane & 15;
  __bf16* stg = stg_all[wv];
  float (*ohc)[16][2] = ohc_all[wv];
  float (*lgb)[2]     = lgb_all[wv];
  __bf16* mystate = p.state + (size_t)wid * STATE_HALVES_PER_WAVE;
  const __bf16* zslot = mystate + 2 * 8 * 2048;           // always-zero h slot

  // pull the swizzled weight image (L2-resident, ~370KB) toward this WGP
  {
    const int totalLines = (HALVES_BIG + HALVES_WI0 + HALVES_WI1 + HALVES_WDEC) / 128;
    for (int t = threadIdx.x; t < totalLines; t += 256)
      __builtin_prefetch(p.wbig + t * 128, 0, 0);
  }

  // stage biases into LDS (shared by the whole block)
  {
    const float* bsrc[6] = { p.bi0, p.bo10, p.bo20, p.bi1, p.bo11, p.bo21 };
#pragma unroll
    for (int a = 0; a < 6; ++a)
      if (threadIdx.x < 128) bias_all[a][threadIdx.x] = bsrc[a][threadIdx.x];
  }

  // zero recurrent state incl. zero-slot (global) and one-hot state (LDS)
  {
    bf16x8 z;
#pragma unroll
    for (int t = 0; t < 8; ++t) z[t] = (__bf16)0.0f;
    bf16x8* sp = (bf16x8*)mystate;
    for (int t = 0; t < STATE_HALVES_PER_WAVE / (8 * 32); ++t) sp[t * 32 + lane] = z;
    float* oc = &ohc[0][0][0];
#pragma unroll
    for (int t = 0; t < 8; ++t) oc[t * 32 + lane] = 0.0f;
  }
  __syncthreads();

  LayerW L0 { p.wbig,             p.wbig + 16384,     p.wbig + 2 * 16384,
              p.wbig + 3 * 16384, p.wbig + 4 * 16384, p.wi0,
              bias_all[0], bias_all[1], bias_all[2] };
  LayerW L1 { p.wbig + 5 * 16384, p.wbig + 6 * 16384, p.wbig + 7 * 16384,
              p.wbig + 8 * 16384, p.wbig + 9 * 16384, p.wi1,
              bias_all[3], bias_all[4], bias_all[5] };

  float logp = 0.0f, phase = 0.0f;
  const float wph00 = p.Wph[0], wph01 = p.Wph[1], wph10 = p.Wph[2], wph11 = p.Wph[3];
  const float bph0 = p.bph[0], bph1 = p.bph[1];
  const float bd0 = p.bdec[0], bd1 = p.bdec[1];
  const bool lo16 = lane < 16;

#pragma unroll 1
  for (int j = 0; j < 8; ++j) {
#pragma unroll 1
    for (int i = 0; i < 8; ++i) {
      const int  nx = (j & 1) ? (7 - i) : i;
      const bool have_left = (i > 0);
      const int  lnx = have_left ? ((j & 1) ? nx + 1 : nx - 1) : nx;

      // x2 = onehot(down) (prev column), x1 = onehot(left) (this column)
      float x2a = ohc[nx][m][0],  x2b = ohc[nx][m][1];
      float x1a = have_left ? ohc[lnx][m][0] : 0.0f;
      float x1b = have_left ? ohc[lnx][m][1] : 0.0f;

      // layer-0 input A fragment: K=0..3 = [x1, x2], rest zero
      bf16x16 catf = zero16();
      catf[0] = (__bf16)(lo16 ? x1a : 0.0f);
      catf[1] = (__bf16)(lo16 ? x1b : 0.0f);
      catf[2] = (__bf16)(lo16 ? x2a : 0.0f);
      catf[3] = (__bf16)(lo16 ? x2b : 0.0f);
      bf16x16 inF0[1] = { catf };

      __bf16* slot0c = mystate + (0 * 8 + nx) * 2048;
      const __bf16* slot0l = have_left ? mystate + (0 * 8 + lnx) * 2048 : zslot;
      run_layer<1>(L0, slot0l, slot0c, inF0, stg, lane);

      // layer-1 input: x (staged) as 4 A fragments + x2 tail fragment (K=128..129)
      bf16x16 inF1[5];
#pragma unroll
      for (int kt = 0; kt < 4; ++kt) inF1[kt] = loadAfragLDS(stg, kt, lane);
      waitds();
      bf16x16 tf = zero16();
      tf[0] = (__bf16)(lo16 ? x2a : 0.0f);
      tf[1] = (__bf16)(lo16 ? x2b : 0.0f);
      inF1[4] = tf;

      __bf16* slot1c = mystate + (1 * 8 + nx) * 2048;
      const __bf16* slot1l = have_left ? mystate + (1 * 8 + lnx) * 2048 : zslot;
      run_layer<5>(L1, slot1l, slot1c, inF1, stg, lane);

      // ---- decoder logits via WMMA: logits(16x2) = x(16x128) @ Wdec(128x16-padded) ----
      v8f dacc;
#pragma unroll
      for (int r = 0; r < 8; ++r) dacc[r] = 0.0f;
      {
        bf16x16 db = loadBfrag(p.wdec, 0, 0, lane);
#pragma unroll
        for (int kt = 0; kt < 4; ++kt) {
          bf16x16 nb = db;
          if (kt + 1 < 4) nb = loadBfrag(p.wdec, kt + 1, 0, lane);
          dacc = wmma_bf16(loadAfragLDS(stg, kt, lane), db, dacc);
          db = nb;
        }
      }
      // scatter valid columns (n = 0,1) to per-row LDS slots
      {
        int nloc = lane & 15;
        int mb   = (lane < 16) ? 0 : 8;
        if (nloc < 2) {
#pragma unroll
          for (int r = 0; r < 8; ++r) lgb[mb + r][nloc] = dacc[r];
        }
      }
      waitds();

      // ---- log-softmax + phase (row-per-lane on lanes 0..15) ----
      if (lo16) {
        float l0 = lgb[m][0] + bd0;
        float l1 = lgb[m][1] + bd1;
        int s = p.samples[(size_t)(m0 + m) * 64 + nx * 8 + j];
        float mx  = fmaxf(l0, l1);
        float lse = mx + __logf(__expf(l0 - mx) + __expf(l1 - mx));
        logp += (s ? l1 : l0) - lse;
        float p0 = l0 * wph00 + l1 * wph10 + bph0;
        float p1 = l0 * wph01 + l1 * wph11 + bph1;
        float pk = s ? p1 : p0;
        phase += 3.14159265358979323846f * (pk / (1.0f + fabsf(pk)));
        ohc[nx][m][0] = s ? 0.0f : 1.0f;
        ohc[nx][m][1] = s ? 1.0f : 0.0f;
      }
      waitds();
    }
  }

  if (lo16) {
    p.out[2 * (m0 + m) + 0] = 0.5f * logp;
    p.out[2 * (m0 + m) + 1] = phase;
  }
}

// ======================= host launcher =======================
extern "C" void kernel_launch(void* const* d_in, const int* in_sizes, int n_in,
                              void* d_out, int out_size, void* d_ws, size_t ws_size,
                              hipStream_t stream) {
  __bf16* wbig  = (__bf16*)d_ws;
  __bf16* wi0   = wbig + HALVES_BIG;
  __bf16* wi1   = wi0 + HALVES_WI0;
  __bf16* wdec  = wi1 + HALVES_WI1;
  __bf16* state = wdec + HALVES_WDEC;

  // d_in order: samples, [U1,U2,Wi,bi,Wh,Wo1,bo1,Wo2,bo2] x2, Wdec, bdec, Wph, bph
  WInit wi;
  wi.big[0] = (const float*)d_in[1];   // U1_0
  wi.big[1] = (const float*)d_in[2];   // U2_0
  wi.big[2] = (const float*)d_in[5];   // Wh_0
  wi.big[3] = (const float*)d_in[6];   // Wo1_0
  wi.big[4] = (const float*)d_in[8];   // Wo2_0
  wi.big[5] = (const float*)d_in[10];  // U1_1
  wi.big[6] = (const float*)d_in[11];  // U2_1
  wi.big[7] = (const float*)d_in[14];  // Wh_1
  wi.big[8] = (const float*)d_in[15];  // Wo1_1
  wi.big[9] = (const float*)d_in[17];  // Wo2_1
  wi.wi0src  = (const float*)d_in[3];
  wi.wi1src  = (const float*)d_in[12];
  wi.wdecsrc = (const float*)d_in[19];
  wi.wbig = wbig; wi.wi0 = wi0; wi.wi1 = wi1; wi.wdec = wdec;

  int tot = HALVES_BIG + HALVES_WI0 + HALVES_WI1 + HALVES_WDEC;
  init_weights_kernel<<<(tot + 255) / 256, 256, 0, stream>>>(wi);

  MainP mp;
  mp.samples = (const int*)d_in[0];
  mp.wbig = wbig; mp.wi0 = wi0; mp.wi1 = wi1; mp.wdec = wdec; mp.state = state;
  mp.bi0  = (const float*)d_in[4];
  mp.bo10 = (const float*)d_in[7];
  mp.bo20 = (const float*)d_in[9];
  mp.bi1  = (const float*)d_in[13];
  mp.bo11 = (const float*)d_in[16];
  mp.bo21 = (const float*)d_in[18];
  mp.bdec = (const float*)d_in[20];
  mp.Wph  = (const float*)d_in[21];
  mp.bph  = (const float*)d_in[22];
  mp.out  = (float*)d_out;

  rnn_kernel<<<NBLOCKS, 256, 0, stream>>>(mp);
}
